// GAT_53635551592822
// MI455X (gfx1250) — compile-verified, matrix operand order checked by reading, not run on hardware
//
#include <hip/hip_runtime.h>
#include <hip/hip_bf16.h>

typedef __attribute__((ext_vector_type(16))) _Float16 v16h;
typedef __attribute__((ext_vector_type(8)))  float    v8f;

#define NEG_SLOPE 0.2f

// ---------------- helpers ----------------
__device__ __forceinline__ float lrelu(float a) { return a > 0.f ? a : NEG_SLOPE * a; }
__device__ __forceinline__ float eluf(float v)  { return v > 0.f ? v : __expf(v) - 1.f; }

// Order-preserving float atomic max via int/uint atomics (init must be -inf).
__device__ __forceinline__ void atomicMaxF32(float* addr, float val) {
  if (val >= 0.f) atomicMax((int*)addr, __float_as_int(val));
  else            atomicMin((unsigned int*)addr, __float_as_uint(val));
}

// ---------------- prep: transpose + f16 conversion of weights ----------------
// W1 [128,128] row-major (k-major) -> W1t [n][k] f16 ; W2 [128,16] -> W2t [16][128] f16
__global__ void prep_weights(const float* __restrict__ W1, const float* __restrict__ W2,
                             _Float16* __restrict__ W1t, _Float16* __restrict__ W2t) {
  int tid = blockIdx.x * blockDim.x + threadIdx.x;
  if (tid < 128 * 128) {
    int n = tid >> 7, k = tid & 127;
    W1t[n * 128 + k] = (_Float16)W1[k * 128 + n];
  }
  int t2 = tid - 128 * 128;
  if (t2 >= 0 && t2 < 16 * 128) {
    int n = t2 >> 7, k = t2 & 127;
    W2t[n * 128 + k] = (_Float16)W2[k * 16 + n];
  }
}

// ---------------- init: zero accumulators, -inf maxes ----------------
__global__ void init_buffers(float* __restrict__ agg1, float* __restrict__ outp,
                             float* __restrict__ amax1, float* __restrict__ denom1,
                             float* __restrict__ amax2, float* __restrict__ denom2, int N) {
  int tid = blockIdx.x * blockDim.x + threadIdx.x;
  if (tid < N * 128) agg1[tid] = 0.f;
  if (tid < N * 16)  outp[tid] = 0.f;
  if (tid < N * 4)  { amax1[tid] = -__builtin_inff(); denom1[tid] = 0.f; }
  if (tid < N)      { amax2[tid] = -__builtin_inff(); denom2[tid] = 0.f; }
}

// ---------------- GEMM1: h1 = x @ W1  (M=N nodes, K=128, Nout=128) ----------------
// Block = 256 threads = 8 waves. Wave w owns 16x16 output tile at (blockIdx.x*16, w*16).
__global__ void __launch_bounds__(256) gemm1_x_w1(const float* __restrict__ x,
                                                  const _Float16* __restrict__ W1t,
                                                  float* __restrict__ h1) {
  const int w  = threadIdx.x >> 5;
  const int l  = threadIdx.x & 31;
  const int lm = l & 15;        // row (A) / col (B) within tile
  const int kh = l >> 4;        // K-half selector
  const int row0 = blockIdx.x * 16;
  const int colb = w * 16;

  const float*    arow = x   + (size_t)(row0 + lm) * 128;
  const _Float16* brow = W1t + (size_t)(colb + lm) * 128;

  v8f acc = {};
#pragma unroll
  for (int kk = 0; kk < 4; ++kk) {
    const int kbase = kk * 32 + kh * 8;
    v16h a, b;
#pragma unroll
    for (int i = 0; i < 8; ++i) {
      a[i]     = (_Float16)arow[kbase + i];
      a[8 + i] = (_Float16)arow[kbase + 16 + i];
      b[i]     = brow[kbase + i];
      b[8 + i] = brow[kbase + 16 + i];
    }
    acc = __builtin_amdgcn_wmma_f32_16x16x32_f16(false, a, false, b, (short)0, acc,
                                                 false, false);
  }
  // D layout: reg r -> row = r + 8*kh, col = lm
  float* orow = h1 + (size_t)(row0 + 8 * kh) * 128 + colb + lm;
#pragma unroll
  for (int r = 0; r < 8; ++r) orow[(size_t)r * 128] = acc[r];
}

// ---------------- GEMM2: h2 = elu(agg1 + b1) @ W2  (K=128, Nout=16) ----------------
// elu+bias fused into A-fragment load. One wave per 16-row tile, 8 tiles/block.
__global__ void __launch_bounds__(256) gemm2_h_w2(const float* __restrict__ agg1,
                                                  const float* __restrict__ b1,
                                                  const _Float16* __restrict__ W2t,
                                                  float* __restrict__ h2, int Mtiles) {
  const int w    = threadIdx.x >> 5;
  const int tile = blockIdx.x * 8 + w;
  if (tile >= Mtiles) return;     // wave-uniform: EXEC stays all-ones
  const int l  = threadIdx.x & 31;
  const int lm = l & 15;
  const int kh = l >> 4;
  const int row0 = tile * 16;

  const float*    arow = agg1 + (size_t)(row0 + lm) * 128;
  const _Float16* brow = W2t  + (size_t)lm * 128;

  v8f acc = {};
#pragma unroll
  for (int kk = 0; kk < 4; ++kk) {
    const int kbase = kk * 32 + kh * 8;
    v16h a, b;
#pragma unroll
    for (int i = 0; i < 8; ++i) {
      a[i]     = (_Float16)eluf(arow[kbase + i]      + b1[kbase + i]);
      a[8 + i] = (_Float16)eluf(arow[kbase + 16 + i] + b1[kbase + 16 + i]);
      b[i]     = brow[kbase + i];
      b[8 + i] = brow[kbase + 16 + i];
    }
    acc = __builtin_amdgcn_wmma_f32_16x16x32_f16(false, a, false, b, (short)0, acc,
                                                 false, false);
  }
  float* orow = h2 + (size_t)(row0 + 8 * kh) * 16 + lm;
#pragma unroll
  for (int r = 0; r < 8; ++r) orow[(size_t)r * 16] = acc[r];
}

// ---------------- per-node attention logits ----------------
__global__ void alpha1_kernel(const float* __restrict__ h1, const float* __restrict__ a_src,
                              const float* __restrict__ a_dst, float* __restrict__ asrc,
                              float* __restrict__ adst, int N) {
  int tid = blockIdx.x * blockDim.x + threadIdx.x;
  if (tid >= N * 4) return;
  int n = tid >> 2, h = tid & 3;
  const float* hp = h1 + (size_t)n * 128 + h * 32;
  const float* as = a_src + h * 32;
  const float* ad = a_dst + h * 32;
  float s = 0.f, d = 0.f;
#pragma unroll
  for (int c = 0; c < 32; ++c) { s += hp[c] * as[c]; d += hp[c] * ad[c]; }
  asrc[tid] = s; adst[tid] = d;
}

__global__ void alpha2_kernel(const float* __restrict__ h2, const float* __restrict__ a_src,
                              const float* __restrict__ a_dst, float* __restrict__ asrc,
                              float* __restrict__ adst, int N) {
  int tid = blockIdx.x * blockDim.x + threadIdx.x;
  if (tid >= N) return;
  const float* hp = h2 + (size_t)tid * 16;
  float s = 0.f, d = 0.f;
#pragma unroll
  for (int c = 0; c < 16; ++c) { s += hp[c] * a_src[c]; d += hp[c] * a_dst[c]; }
  asrc[tid] = s; adst[tid] = d;
}

// ---------------- edge decode (implicit self-loops appended) ----------------
__device__ __forceinline__ void edge_sd(const int* __restrict__ ei, int E, int e,
                                        int& s, int& d) {
  if (e < E) { s = ei[e]; d = ei[E + e]; } else { s = d = e - E; }
}

// ---------------- layer-1 edge passes (H=4) ----------------
__global__ void edge_max1(const int* __restrict__ ei, int E, int N,
                          const float* __restrict__ asrc, const float* __restrict__ adst,
                          float* __restrict__ amax) {
  int e = blockIdx.x * blockDim.x + threadIdx.x;
  if (e >= E + N) return;
  int s, d; edge_sd(ei, E, e, s, d);
#pragma unroll
  for (int h = 0; h < 4; ++h)
    atomicMaxF32(&amax[d * 4 + h], lrelu(asrc[s * 4 + h] + adst[d * 4 + h]));
}

__global__ void edge_sum1(const int* __restrict__ ei, int E, int N,
                          const float* __restrict__ asrc, const float* __restrict__ adst,
                          const float* __restrict__ amax, float* __restrict__ denom) {
  int e = blockIdx.x * blockDim.x + threadIdx.x;
  if (e >= E + N) return;
  int s, d; edge_sd(ei, E, e, s, d);
#pragma unroll
  for (int h = 0; h < 4; ++h) {
    float a = lrelu(asrc[s * 4 + h] + adst[d * 4 + h]);
    atomicAdd(&denom[d * 4 + h], __expf(a - amax[d * 4 + h]));
  }
}

__global__ void edge_agg1(const int* __restrict__ ei, int E, int N,
                          const float* __restrict__ asrc, const float* __restrict__ adst,
                          const float* __restrict__ amax, const float* __restrict__ denom,
                          const float* __restrict__ h1, float* __restrict__ agg) {
  long long tid = (long long)blockIdx.x * blockDim.x + threadIdx.x;
  if (tid >= (long long)(E + N) * 128) return;
  int e = (int)(tid >> 7);
  int c = (int)(tid & 127);
  int h = c >> 5;
  int s, d; edge_sd(ei, E, e, s, d);
  float a    = lrelu(asrc[s * 4 + h] + adst[d * 4 + h]);
  float attn = __expf(a - amax[d * 4 + h]) / denom[d * 4 + h];
  atomicAdd(&agg[(size_t)d * 128 + c], h1[(size_t)s * 128 + c] * attn);
}

// ---------------- layer-2 edge passes (H=1, C=16) ----------------
__global__ void edge_max2(const int* __restrict__ ei, int E, int N,
                          const float* __restrict__ asrc, const float* __restrict__ adst,
                          float* __restrict__ amax) {
  int e = blockIdx.x * blockDim.x + threadIdx.x;
  if (e >= E + N) return;
  int s, d; edge_sd(ei, E, e, s, d);
  atomicMaxF32(&amax[d], lrelu(asrc[s] + adst[d]));
}

__global__ void edge_sum2(const int* __restrict__ ei, int E, int N,
                          const float* __restrict__ asrc, const float* __restrict__ adst,
                          const float* __restrict__ amax, float* __restrict__ denom) {
  int e = blockIdx.x * blockDim.x + threadIdx.x;
  if (e >= E + N) return;
  int s, d; edge_sd(ei, E, e, s, d);
  float a = lrelu(asrc[s] + adst[d]);
  atomicAdd(&denom[d], __expf(a - amax[d]));
}

__global__ void edge_agg2(const int* __restrict__ ei, int E, int N,
                          const float* __restrict__ asrc, const float* __restrict__ adst,
                          const float* __restrict__ amax, const float* __restrict__ denom,
                          const float* __restrict__ h2, float* __restrict__ outp) {
  long long tid = (long long)blockIdx.x * blockDim.x + threadIdx.x;
  if (tid >= (long long)(E + N) * 16) return;
  int e = (int)(tid >> 4);
  int c = (int)(tid & 15);
  int s, d; edge_sd(ei, E, e, s, d);
  float a    = lrelu(asrc[s] + adst[d]);
  float attn = __expf(a - amax[d]) / denom[d];
  atomicAdd(&outp[(size_t)d * 16 + c], h2[(size_t)s * 16 + c] * attn);
}

__global__ void final_bias(float* __restrict__ outp, const float* __restrict__ b2, int N) {
  int tid = blockIdx.x * blockDim.x + threadIdx.x;
  if (tid < N * 16) outp[tid] += b2[tid & 15];
}

// ---------------- launcher ----------------
extern "C" void kernel_launch(void* const* d_in, const int* in_sizes, int n_in,
                              void* d_out, int out_size, void* d_ws, size_t ws_size,
                              hipStream_t stream) {
  const float* x      = (const float*)d_in[0];
  const int*   ei     = (const int*)d_in[1];
  const float* W1     = (const float*)d_in[2];
  const float* a_src1 = (const float*)d_in[3];
  const float* a_dst1 = (const float*)d_in[4];
  const float* b1     = (const float*)d_in[5];
  const float* W2     = (const float*)d_in[6];
  const float* a_src2 = (const float*)d_in[7];
  const float* a_dst2 = (const float*)d_in[8];
  const float* b2     = (const float*)d_in[9];
  float* outp = (float*)d_out;

  const int N    = in_sizes[0] / 128;
  const int E    = in_sizes[1] / 2;
  const int Etot = E + N;

  // workspace carve-out (256B aligned)
  char* p = (char*)d_ws;
  auto alloc = [&](size_t bytes) -> char* {
    char* r = p; p += (bytes + 255) & ~(size_t)255; return r;
  };
  _Float16* W1t    = (_Float16*)alloc(128 * 128 * 2);
  _Float16* W2t    = (_Float16*)alloc(16 * 128 * 2);
  float*    h1     = (float*)alloc((size_t)N * 128 * 4);
  float*    agg1   = (float*)alloc((size_t)N * 128 * 4);
  float*    h2     = (float*)alloc((size_t)N * 16 * 4);
  float*    asrc1  = (float*)alloc((size_t)N * 4 * 4);
  float*    adst1  = (float*)alloc((size_t)N * 4 * 4);
  float*    amax1  = (float*)alloc((size_t)N * 4 * 4);
  float*    denom1 = (float*)alloc((size_t)N * 4 * 4);
  float*    asrc2  = (float*)alloc((size_t)N * 4);
  float*    adst2  = (float*)alloc((size_t)N * 4);
  float*    amax2  = (float*)alloc((size_t)N * 4);
  float*    denom2 = (float*)alloc((size_t)N * 4);

  const int T = 256;
  auto blk = [](long long n, int t) { return (unsigned)((n + t - 1) / t); };

  prep_weights<<<blk(128 * 128 + 16 * 128, T), T, 0, stream>>>(W1, W2, W1t, W2t);
  init_buffers<<<blk((long long)N * 128, T), T, 0, stream>>>(agg1, outp, amax1, denom1,
                                                             amax2, denom2, N);
  // ---- layer 1 ----
  gemm1_x_w1<<<N / 16, T, 0, stream>>>(x, W1t, h1);
  alpha1_kernel<<<blk((long long)N * 4, T), T, 0, stream>>>(h1, a_src1, a_dst1,
                                                            asrc1, adst1, N);
  edge_max1<<<blk(Etot, T), T, 0, stream>>>(ei, E, N, asrc1, adst1, amax1);
  edge_sum1<<<blk(Etot, T), T, 0, stream>>>(ei, E, N, asrc1, adst1, amax1, denom1);
  edge_agg1<<<blk((long long)Etot * 128, T), T, 0, stream>>>(ei, E, N, asrc1, adst1,
                                                             amax1, denom1, h1, agg1);
  // ---- layer 2 (elu + b1 fused into GEMM2 A-load) ----
  const int Mtiles = N / 16;
  gemm2_h_w2<<<blk(Mtiles, 8), T, 0, stream>>>(agg1, b1, W2t, h2, Mtiles);
  alpha2_kernel<<<blk(N, T), T, 0, stream>>>(h2, a_src2, a_dst2, asrc2, adst2, N);
  edge_max2<<<blk(Etot, T), T, 0, stream>>>(ei, E, N, asrc2, adst2, amax2);
  edge_sum2<<<blk(Etot, T), T, 0, stream>>>(ei, E, N, asrc2, adst2, amax2, denom2);
  edge_agg2<<<blk((long long)Etot * 16, T), T, 0, stream>>>(ei, E, N, asrc2, adst2,
                                                            amax2, denom2, h2, outp);
  final_bias<<<blk((long long)N * 16, T), T, 0, stream>>>(outp, b2, N);
}
